// GatingNetwork_34694745817706
// MI455X (gfx1250) — matrix operand order repaired and checked by reading, not compiled, not run on hardware
//
#include <hip/hip_runtime.h>

#define N_MODELS 8
#define HIDDEN   32
#define EPS      1e-8f

typedef __attribute__((ext_vector_type(16))) _Float16 v16h;
typedef __attribute__((ext_vector_type(8)))  float    v8f;

// One wave32 handles 32 rows (two 16-row WMMA tiles).
// Block = 256 threads = 8 waves = 256 rows.
__global__ __launch_bounds__(256) void gating_wmma_kernel(
    const float* __restrict__ probs,   // [B, 8]
    const float* __restrict__ W1,      // [8, 32] row-major
    const float* __restrict__ b1,      // [32]
    const float* __restrict__ W2,      // [32, 8] row-major
    const float* __restrict__ b2,      // [8]
    float*       __restrict__ out_w,   // [B, 8]
    int*         __restrict__ out_idx) // [B, 2]
{
    __shared__ float lds_h[8][16 * HIDDEN];        // per-wave 16x32 hidden tile
    __shared__ float lds_logit[8][32 * N_MODELS];  // per-wave 32x8 logits

    const int tid  = threadIdx.x;
    const int wave = tid >> 5;
    const int lane = tid & 31;
    const int hi   = lane >> 4;   // 0: lanes 0-15, 1: lanes 16-31
    const int n    = lane & 15;   // column / row-in-halfwave index

    // ---- Build f16 B fragments (32x16 K-major layout; K padded with zeros) ----
    // B layout (16-bit, 32x16): lanes 0-15 hold K=0..7,16..23 (2 per VGPR), lanes 16-31 K=8..15,24..31; N = lane%16.
    v16h bW1_0 = {}, bW1_1 = {}, bW2 = {};
    if (!hi) {
        #pragma unroll
        for (int k = 0; k < N_MODELS; ++k) {           // K = 0..7 valid, rest zero
            bW1_0[k] = (_Float16)W1[k * HIDDEN + n];
            bW1_1[k] = (_Float16)W1[k * HIDDEN + 16 + n];
        }
    }
    if (n < N_MODELS) {                                // N = 0..7 valid, rest zero
        const int kb = hi ? 8 : 0;
        #pragma unroll
        for (int k = 0; k < 16; ++k)
            bW2[k] = (_Float16)W2[(kb + (k < 8 ? k : k + 8)) * N_MODELS + n];
    }

    // ---- Fold biases into WMMA C operands.
    // f32 C/D layout: element (row = r + 8*hi, col = lane&15) lives in VGPR r,
    // so a column-only bias is a broadcast across all 8 C registers.
    const float b1lo = b1[n];
    const float b1hi = b1[16 + n];
    const float b2n  = (n < N_MODELS) ? b2[n] : 0.0f;
    v8f cB1lo, cB1hi, cB2;
    #pragma unroll
    for (int r = 0; r < 8; ++r) { cB1lo[r] = b1lo; cB1hi[r] = b1hi; cB2[r] = b2n; }

    const long long waveRow = ((long long)blockIdx.x * 8 + wave) * 32;

    #pragma unroll
    for (int tile = 0; tile < 2; ++tile) {
        const long long rowBase = waveRow + (long long)tile * 16;

        // ---- A fragment for GEMM1: 16 rows x K32 (K=0..7 = probs, rest zero).
        // A layout: lanes 0-15 elems 0..7 = K0..7; all other positions zero.
        v16h a = {};
        if (!hi) {
            const float4* rp = (const float4*)(probs + (rowBase + n) * N_MODELS);
            if (tile == 0)
                __builtin_prefetch(probs + (rowBase + 16 + n) * N_MODELS, 0, 1);
            float4 p0 = rp[0];
            float4 p1 = rp[1];
            a[0] = (_Float16)p0.x; a[1] = (_Float16)p0.y;
            a[2] = (_Float16)p0.z; a[3] = (_Float16)p0.w;
            a[4] = (_Float16)p1.x; a[5] = (_Float16)p1.y;
            a[6] = (_Float16)p1.z; a[7] = (_Float16)p1.w;
        }

        // GEMM1: [16x32] = [16x8] @ [8x32] via two N-halves; bias pre-loaded in C
        v8f d0 = __builtin_amdgcn_wmma_f32_16x16x32_f16(false, a, false, bW1_0, (short)0, cB1lo, false, false);
        v8f d1 = __builtin_amdgcn_wmma_f32_16x16x32_f16(false, a, false, bW1_1, (short)0, cB1hi, false, false);

        // ReLU, spill h tile to LDS (C/D layout: row = r + 8*hi, col = n)
        #pragma unroll
        for (int r = 0; r < 8; ++r) {
            const int row = r + 8 * hi;
            lds_h[wave][row * HIDDEN + n]      = fmaxf(d0[r], 0.0f);
            lds_h[wave][row * HIDDEN + 16 + n] = fmaxf(d1[r], 0.0f);
        }
        asm volatile("s_wait_dscnt 0" ::: "memory"); // same-wave LDS RAW fence

        // ---- A fragment for GEMM2: 16x32 f16 from LDS h tile
        // lanes 0-15: elems 0..7 = K0..7, elems 8..15 = K16..23
        // lanes 16-31: elems 0..7 = K8..15, elems 8..15 = K24..31
        v16h a2;
        {
            const int kb = hi ? 8 : 0;
            const float4* hp = (const float4*)&lds_h[wave][n * HIDDEN + kb];
            float4 h0 = hp[0], h1 = hp[1], h2 = hp[4], h3 = hp[5];
            v16h t;
            t[0]  = (_Float16)h0.x; t[1]  = (_Float16)h0.y; t[2]  = (_Float16)h0.z; t[3]  = (_Float16)h0.w;
            t[4]  = (_Float16)h1.x; t[5]  = (_Float16)h1.y; t[6]  = (_Float16)h1.z; t[7]  = (_Float16)h1.w;
            t[8]  = (_Float16)h2.x; t[9]  = (_Float16)h2.y; t[10] = (_Float16)h2.z; t[11] = (_Float16)h2.w;
            t[12] = (_Float16)h3.x; t[13] = (_Float16)h3.y; t[14] = (_Float16)h3.z; t[15] = (_Float16)h3.w;
            a2 = t;
        }

        // GEMM2: [16x8] = [16x32] @ [32x8] (N padded to 16); b2 pre-loaded in C
        v8f d2 = __builtin_amdgcn_wmma_f32_16x16x32_f16(false, a2, false, bW2, (short)0, cB2, false, false);

        if (n < N_MODELS) {
            #pragma unroll
            for (int r = 0; r < 8; ++r) {
                const int row = r + 8 * hi;
                lds_logit[wave][(tile * 16 + row) * N_MODELS + n] = d2[r];
            }
        }
        asm volatile("s_wait_dscnt 0" ::: "memory");
    }

    // ---- Epilogue: each lane owns one of the wave's 32 rows ----
    const long long myRow = waveRow + lane;
    float w[8];
    {
        const float4* lp = (const float4*)&lds_logit[wave][lane * N_MODELS];
        float4 l0 = lp[0], l1 = lp[1];
        w[0] = l0.x; w[1] = l0.y; w[2] = l0.z; w[3] = l0.w;
        w[4] = l1.x; w[5] = l1.y; w[6] = l1.z; w[7] = l1.w;
    }
    // softmax
    float m = w[0];
    #pragma unroll
    for (int j = 1; j < 8; ++j) m = fmaxf(m, w[j]);
    float s = 0.0f;
    #pragma unroll
    for (int j = 0; j < 8; ++j) { w[j] = __expf(w[j] - m); s += w[j]; }
    const float invs = 1.0f / s;
    #pragma unroll
    for (int j = 0; j < 8; ++j) w[j] *= invs;

    // top-2 (first occurrence wins ties, like lax.top_k)
    int i1 = 0; float v1 = w[0];
    #pragma unroll
    for (int j = 1; j < 8; ++j) if (w[j] > v1) { v1 = w[j]; i1 = j; }
    int i2 = -1; float v2 = -3.4e38f;
    #pragma unroll
    for (int j = 0; j < 8; ++j) if (j != i1 && w[j] > v2) { v2 = w[j]; i2 = j; }

    const float dnm = 1.0f / (v1 + v2 + EPS);
    float o[8];
    #pragma unroll
    for (int j = 0; j < 8; ++j) o[j] = (j == i1 || j == i2) ? w[j] * dnm : 0.0f;

    float4* op = (float4*)(out_w + myRow * N_MODELS);
    float4 o0, o1;
    o0.x = o[0]; o0.y = o[1]; o0.z = o[2]; o0.w = o[3];
    o1.x = o[4]; o1.y = o[5]; o1.z = o[6]; o1.w = o[7];
    op[0] = o0;
    op[1] = o1;
    int2 iv; iv.x = i1; iv.y = i2;
    ((int2*)out_idx)[myRow] = iv;
}

extern "C" void kernel_launch(void* const* d_in, const int* in_sizes, int n_in,
                              void* d_out, int out_size, void* d_ws, size_t ws_size,
                              hipStream_t stream) {
    const float* probs = (const float*)d_in[0];
    const float* W1    = (const float*)d_in[1];
    const float* b1    = (const float*)d_in[2];
    const float* W2    = (const float*)d_in[3];
    const float* b2    = (const float*)d_in[4];

    const long long B = (long long)in_sizes[0] / N_MODELS; // 4,194,304
    float* out_w  = (float*)d_out;
    int*   out_ix = (int*)(out_w + B * N_MODELS); // tuple: [B*8 f32 | B*2 i32]

    const int rowsPerBlock = 256; // 8 waves x 32 rows
    const int blocks = (int)(B / rowsPerBlock);
    gating_wmma_kernel<<<blocks, 256, 0, stream>>>(probs, W1, b1, W2, b2, out_w, out_ix);
}
